// globalatt_89163521065535
// MI455X (gfx1250) — matrix operand order repaired and checked by reading, not compiled, not run on hardware
//
#include <hip/hip_runtime.h>
#include <math.h>

// ---------------------------------------------------------------------------
// Problem constants (from reference): B=2, C=128, H=W=256, K_SVD=32, KNN=8,
// C_RED=32, GRP=4, P=H*W=65536, M=B*C=256.
// ---------------------------------------------------------------------------

typedef __attribute__((ext_vector_type(16))) __bf16 v16bf;
typedef __attribute__((ext_vector_type(8)))  __bf16 v8bf;
typedef __attribute__((ext_vector_type(8)))  float  v8f;

// ---------------------------------------------------------------------------
// Kernel 1: X_node[b,c,h] = mean_w x[b,c,h,w].  One wave per row (coalesced
// float4 loads, shuffle reduction).  rows = B*C*H = 65536, grid = 8192 x 256.
// ---------------------------------------------------------------------------
__global__ __launch_bounds__(256) void k_rowmean(const float* __restrict__ x,
                                                 float* __restrict__ Xn) {
  const int row  = (blockIdx.x << 3) + (threadIdx.x >> 5);
  const int lane = threadIdx.x & 31;
  const float* xp = x + (size_t)row * 256;
  float4 a = *(const float4*)(xp + lane * 4);
  float4 b = *(const float4*)(xp + 128 + lane * 4);
  float s = a.x + a.y + a.z + a.w + b.x + b.y + b.z + b.w;
  #pragma unroll
  for (int o = 16; o > 0; o >>= 1) s += __shfl_down(s, o, 32);
  if (lane == 0) Xn[row] = s * (1.0f / 256.0f);
}

// ---------------------------------------------------------------------------
// Deterministic block reduction helper (256 threads).
// ---------------------------------------------------------------------------
__device__ __forceinline__ float blk_red(float* red, int t, float v, bool domax) {
  __syncthreads();                // protect any previous use of red[]
  red[t] = v;
  __syncthreads();
  #pragma unroll
  for (int s = 128; s > 0; s >>= 1) {
    if (t < s) red[t] = domax ? fmaxf(red[t], red[t + s]) : (red[t] + red[t + s]);
    __syncthreads();
  }
  return red[0];
}

// ---------------------------------------------------------------------------
// Kernel 2: all "small" math.  One block per batch, 256 threads.
// Produces: att_row (B,H,W), att_col (B,H,W), tcoef (B,C,H), coefx (B*C),
//           a_row (B,H), a_col (B,W)   -- all in workspace.
// ---------------------------------------------------------------------------
__global__ __launch_bounds__(256) void k_small(
    const float* __restrict__ U, const float* __restrict__ S,
    const float* __restrict__ V, const float* __restrict__ w_row,
    const float* __restrict__ w_col, const float* __restrict__ boast,
    const float* __restrict__ res_scale, const float* __restrict__ reduce_w,
    const float* __restrict__ gcn_w, const float* __restrict__ gcn_b,
    const float* __restrict__ expand_w, const float* __restrict__ Xn,
    float* __restrict__ attR, float* __restrict__ attC, float* __restrict__ tco,
    float* __restrict__ cox, float* __restrict__ arw, float* __restrict__ acl) {
  __shared__ float bufA[256 * 32];     // 32KB: Ukn -> support -> out_nodes
  __shared__ float gw[32 * 32];        // gcn_w
  __shared__ float ews[128], rws[128], gbs[32];
  __shared__ float usum_s[256], dinv_s[256], tm_s[256], cml_s[256], tmn_s[256];
  __shared__ float onsum_s[32];
  __shared__ float zx_s[128], ztopo_s[128], zk_s[32], sm_s[128], chan_s[128];
  __shared__ float red_s[256];
  __shared__ float sc[8];

  const int b = blockIdx.x;
  const int t = threadIdx.x;

  if (t < 32) gbs[t] = gcn_b[t];
  for (int i = t; i < 1024; i += 256) gw[i] = gcn_w[i];
  if (t < 128) { ews[t] = expand_w[t]; rws[t] = reduce_w[t]; }
  __syncthreads();

  // --- phase 1: normalized U rows + Usum (thread t = h) ---
  float uk[32];
  float us = 0.f, nn = 0.f;
  const float* Up = U + ((size_t)b * 256 + t) * 32;
  #pragma unroll
  for (int k = 0; k < 32; ++k) { float v = Up[k]; uk[k] = v; us += v; nn += v * v; }
  usum_s[t] = us;
  const float inv = 1.0f / fmaxf(sqrtf(nn), 1e-12f);
  #pragma unroll
  for (int k = 0; k < 32; ++k) { uk[k] *= inv; bufA[t * 32 + k] = uk[k]; }
  __syncthreads();

  // --- phase 2: similarity row + top-8 (stable, descending) ---
  float vals[8]; int idx8[8];
  #pragma unroll
  for (int i = 0; i < 8; ++i) { vals[i] = -3.4e38f; idx8[i] = 0; }
  for (int h2 = 0; h2 < 256; ++h2) {
    const float* rp = &bufA[h2 * 32];
    float s = 0.f;
    #pragma unroll
    for (int k = 0; k < 32; ++k) s += uk[k] * rp[k];
    if (s > vals[7]) {
      int j = 7;
      while (j > 0 && s > vals[j - 1]) { vals[j] = vals[j - 1]; idx8[j] = idx8[j - 1]; --j; }
      vals[j] = s; idx8[j] = h2;
    }
  }
  float deg = 0.f;
  #pragma unroll
  for (int i = 0; i < 8; ++i) deg += vals[i];
  dinv_s[t] = 1.0f / sqrtf(deg);
  __syncthreads();

  // --- phase 3: support rows overwrite bufA; z_x ---
  const float* Xb = Xn + (size_t)b * 32768;
  {
    float xr[32];
    #pragma unroll
    for (int o = 0; o < 32; ++o) {
      float a = 0.f;
      #pragma unroll
      for (int g = 0; g < 4; ++g) a += Xb[(o * 4 + g) * 256 + t] * rws[o * 4 + g];
      xr[o] = a;
    }
    #pragma unroll
    for (int d = 0; d < 32; ++d) {
      float a = 0.f;
      #pragma unroll
      for (int o = 0; o < 32; ++o) a += xr[o] * gw[o * 32 + d];
      bufA[t * 32 + d] = a;
    }
  }
  if (t < 128) {
    float a = 0.f;
    for (int h = 0; h < 256; ++h) a += Xb[t * 256 + h];
    zx_s[t] = a * (1.0f / 256.0f);
  }
  __syncthreads();

  // --- phase 4: KNN aggregate -> out_nodes (registers), tm, cml ---
  float on[32];
  #pragma unroll
  for (int d = 0; d < 32; ++d) on[d] = gbs[d];
  const float dh = dinv_s[t];
  #pragma unroll
  for (int i = 0; i < 8; ++i) {
    const float wv = vals[i] * dh * dinv_s[idx8[i]];
    const float* sp = &bufA[idx8[i] * 32];
    #pragma unroll
    for (int d = 0; d < 32; ++d) on[d] += wv * sp[d];
  }
  #pragma unroll
  for (int d = 0; d < 32; ++d) on[d] = fmaxf(on[d], 0.f);
  float tmv = 0.f, cmv = 0.f;
  #pragma unroll
  for (int c = 0; c < 128; ++c) {
    const float e = on[c >> 2] * ews[c];
    tmv += fabsf(e); cmv += e;
  }
  tm_s[t] = tmv; cml_s[t] = cmv;
  __syncthreads();                 // agg reads of bufA complete

  // deterministic onsum: write out_nodes rows into bufA, column-sum
  #pragma unroll
  for (int d = 0; d < 32; ++d) bufA[t * 32 + d] = on[d];
  __syncthreads();
  if (t < 32) {
    float a = 0.f;
    for (int h = 0; h < 256; ++h) a += bufA[h * 32 + t];
    onsum_s[t] = a;
  }
  __syncthreads();

  // --- phase 5: reductions (tmax, weight sums), z_topo ---
  const float tmax_raw = blk_red(red_s, t, tmv, true);
  const float swr = blk_red(red_s, t, w_row[t], false);
  const float swc = blk_red(red_s, t, w_col[t], false);
  if (t == 0) {
    float ss = 0.f;
    for (int k = 0; k < 32; ++k) ss += S[b * 32 + k];
    sc[4] = ss; sc[3] = ss + swr + swc;
  }
  if (t < 128) ztopo_s[t] = ews[t] * onsum_s[t >> 2] * (1.0f / 256.0f);
  __syncthreads();

  // --- phase 6: att_row (softmax over w), tmn ---
  {
    const float tmn = tm_s[t] / fmaxf(tmax_raw, 1e-6f);
    tmn_s[t] = tmn;
    const float base = 0.5f * tmn, usv = usum_s[t];
    float mx = -3.4e38f;
    for (int w = 0; w < 256; ++w) mx = fmaxf(mx, usv * w_row[w] + base);
    float den = 0.f;
    for (int w = 0; w < 256; ++w) den += expf(usv * w_row[w] + base - mx);
    const float idn = 1.0f / den;
    float* arp = attR + ((size_t)b * 256 + t) * 256;
    for (int w = 0; w < 256; ++w) arp[w] = expf(usv * w_row[w] + base - mx) * idn;
  }
  __syncthreads();

  // --- phase 7: att_col (softmax over h), thread t = w ---
  {
    const float wc = w_col[t];
    float mx = -3.4e38f;
    for (int h = 0; h < 256; ++h) mx = fmaxf(mx, usum_s[h] * wc + 0.5f * tmn_s[h]);
    float den = 0.f;
    for (int h = 0; h < 256; ++h) den += expf(usum_s[h] * wc + 0.5f * tmn_s[h] - mx);
    const float idn = 1.0f / den;
    for (int h = 0; h < 256; ++h)
      attC[((size_t)b * 256 + h) * 256 + t] =
          expf(usum_s[h] * wc + 0.5f * tmn_s[h] - mx) * idn;
  }

  // --- phase 8: z_k ---
  if (t < 32) {
    float a = 0.f;
    for (int c = 0; c < 128; ++c)
      a += (zx_s[c] + 0.5f * ztopo_s[c]) * V[((size_t)b * 128 + c) * 32 + t];
    zk_s[t] = a;
  }
  __syncthreads();
  if (t == 0) { float zs = 0.f; for (int k = 0; k < 32; ++k) zs += zk_s[k]; sc[2] = zs; }
  __syncthreads();

  // --- phase 9: w4 softmax over 640 -> a_row / a_col / a_chan ---
  const float tot = sc[3];
  float lm = -3.4e38f;
  for (int i = t; i < 640; i += 256) lm = fmaxf(lm, tot * boast[i]);
  const float m4 = blk_red(red_s, t, lm, true);
  float ls = 0.f;
  for (int i = t; i < 640; i += 256) ls += expf(tot * boast[i] - m4);
  const float d4 = blk_red(red_s, t, ls, false);
  const float id4 = 1.0f / d4;
  for (int i = t; i < 640; i += 256) {
    const float v = expf(tot * boast[i] - m4) * id4;
    if (i < 256)      arw[b * 256 + i] = v;
    else if (i < 512) acl[b * 256 + (i - 256)] = v;
    else              chan_s[i - 512] = v;
  }

  // --- phase 10: sm (softmax_c of z_topo), ch_rec, coefx ---
  const float zt = (t < 128) ? ztopo_s[t] : -3.4e38f;
  const float mzt = blk_red(red_s, t, zt, true);
  const float ez = (t < 128) ? expf(ztopo_s[t] - mzt) : 0.f;
  const float dz = blk_red(red_s, t, ez, false);
  if (t < 128) {
    sm_s[t] = ez / dz;
    const float* Vp = V + ((size_t)b * 128 + t) * 32;
    float vz = 0.f, vs = 0.f;
    #pragma unroll
    for (int k = 0; k < 32; ++k) { const float vv = Vp[k]; vz += vv * zk_s[k]; vs += vv * S[b * 32 + k]; }
    const float chr = (vz * sc[4] + sc[2] * vs) * (1.0f / 64.0f);
    const float sg = 1.0f / (1.0f + expf(-chr));
    cox[b * 128 + t] = res_scale[0] + chan_s[t] * sg;
  }
  __syncthreads();

  // --- phase 11: tcoef[b,c,h] = (sm[c] + sigmoid(mean_c exp)) * exp[c,h] ---
  const float sgh = 1.0f / (1.0f + expf(-cml_s[t] * (1.0f / 128.0f)));
  #pragma unroll
  for (int c = 0; c < 128; ++c) {
    const float e = on[c >> 2] * ews[c];
    tco[((size_t)b * 128 + c) * 256 + t] = (sm_s[c] + sgh) * e;
  }
}

// ---------------------------------------------------------------------------
// Kernel 3: R[b,c,h] = dot(att_row[b,h,:], x[b,c,h,:]) -> bf16.  Wave/row.
// ---------------------------------------------------------------------------
__global__ __launch_bounds__(256) void k_R(const float* __restrict__ x,
                                           const float* __restrict__ attR,
                                           __bf16* __restrict__ Rbf) {
  const int row  = (blockIdx.x << 3) + (threadIdx.x >> 5);  // row = m*256 + h
  const int lane = threadIdx.x & 31;
  const int h = row & 255;
  const int b = row >> 15;
  const float* xp = x + (size_t)row * 256;
  const float* ap = attR + ((size_t)(b * 256 + h)) * 256;
  float4 xa = *(const float4*)(xp + lane * 4);
  float4 xb = *(const float4*)(xp + 128 + lane * 4);
  float4 aa = *(const float4*)(ap + lane * 4);
  float4 ab = *(const float4*)(ap + 128 + lane * 4);
  float s = xa.x * aa.x + xa.y * aa.y + xa.z * aa.z + xa.w * aa.w +
            xb.x * ab.x + xb.y * ab.y + xb.z * ab.z + xb.w * ab.w;
  #pragma unroll
  for (int o = 16; o > 0; o >>= 1) s += __shfl_down(s, o, 32);
  if (lane == 0) Rbf[row] = (__bf16)s;
}

// ---------------------------------------------------------------------------
// Kernel 4: Cctx[b,c,w] = sum_h att_col[b,h,w]*x[b,c,h,w] -> bf16.
// One block per (b,c), thread = w (coalesced column reduction).
// ---------------------------------------------------------------------------
__global__ __launch_bounds__(256) void k_Cctx(const float* __restrict__ x,
                                              const float* __restrict__ attC,
                                              __bf16* __restrict__ Cbf) {
  const int bc = blockIdx.x;
  const int b  = bc >> 7;
  const int w  = threadIdx.x;
  const float* xp = x + (size_t)bc * 65536 + w;
  const float* ap = attC + (size_t)b * 65536 + w;
  float acc = 0.f;
  #pragma unroll 4
  for (int h = 0; h < 256; ++h) acc += ap[h * 256] * xp[h * 256];
  Cbf[bc * 256 + w] = (__bf16)acc;
}

// ---------------------------------------------------------------------------
// Kernel 5: fused dual-GEMM (R@G^T and Cctx@G^T, K=256, bf16 WMMA, f32 acc)
// + full epilogue.  One wave per block; each wave owns a 16-column G tile
// (converted f32->bf16 in registers, contiguous-k B-layout) and loops 4 M-tiles.
//   D layout: VGPR r <-> M = m0 + r + 8*half;  lane%16 <-> N (= p).
// ---------------------------------------------------------------------------
__global__ __launch_bounds__(32) void k_gemm(
    const float* __restrict__ G, const float* __restrict__ x,
    const __bf16* __restrict__ Rbf, const __bf16* __restrict__ Cbf,
    const float* __restrict__ cox, const float* __restrict__ arw,
    const float* __restrict__ acl, const float* __restrict__ tco,
    float* __restrict__ out) {
  const int lane = threadIdx.x & 31;
  const int l    = lane & 15;
  const int half = lane >> 4;
  const int n0   = blockIdx.x * 16;
  const int h    = n0 >> 8;          // constant across the 16-wide tile
  const int w    = (n0 & 255) + l;
  const int p    = n0 + l;

  // B fragments: column N = p, K elems {half*16 .. half*16+15} per k-step.
  v16bf bfrag[8];
  const float* gq = G + (size_t)p * 256 + half * 16;
  #pragma unroll
  for (int ks = 0; ks < 8; ++ks) {
    const float4* gp = (const float4*)(gq + ks * 32);
    float4 q0 = gp[0], q1 = gp[1], q2 = gp[2], q3 = gp[3];
    v16bf bv;
    bv[0]  = (__bf16)q0.x; bv[1]  = (__bf16)q0.y; bv[2]  = (__bf16)q0.z; bv[3]  = (__bf16)q0.w;
    bv[4]  = (__bf16)q1.x; bv[5]  = (__bf16)q1.y; bv[6]  = (__bf16)q1.z; bv[7]  = (__bf16)q1.w;
    bv[8]  = (__bf16)q2.x; bv[9]  = (__bf16)q2.y; bv[10] = (__bf16)q2.z; bv[11] = (__bf16)q2.w;
    bv[12] = (__bf16)q3.x; bv[13] = (__bf16)q3.y; bv[14] = (__bf16)q3.z; bv[15] = (__bf16)q3.w;
    bfrag[ks] = bv;
  }

  #pragma unroll
  for (int i = 0; i < 4; ++i) {
    const int m0   = (blockIdx.y * 4 + i) * 16;
    const int b    = m0 >> 7;
    const int rowa = m0 + l;                       // A row for this lane
    v8f accR = {};
    v8f accC = {};
    #pragma unroll
    for (int ks = 0; ks < 8; ++ks) {
      // A layout: half0 -> K{k0..k0+15}; half1 -> K{k0+16..k0+31}.
      const int kb = ks * 32 + half * 16;
      const v8bf* rA = (const v8bf*)(Rbf + (size_t)rowa * 256 + kb);
      const v8bf* cA = (const v8bf*)(Cbf + (size_t)rowa * 256 + kb);
      v16bf aR = __builtin_shufflevector(rA[0], rA[1], 0, 1, 2, 3, 4, 5, 6, 7,
                                         8, 9, 10, 11, 12, 13, 14, 15);
      v16bf aC = __builtin_shufflevector(cA[0], cA[1], 0, 1, 2, 3, 4, 5, 6, 7,
                                         8, 9, 10, 11, 12, 13, 14, 15);
      accR = __builtin_amdgcn_wmma_f32_16x16x32_bf16(false, aR, false, bfrag[ks],
                                                     (short)0, accR, false, false);
      accC = __builtin_amdgcn_wmma_f32_16x16x32_bf16(false, aC, false, bfrag[ks],
                                                     (short)0, accC, false, false);
    }
    // Fused epilogue: out = coefx*x + a_row*RO + a_col*CO + a_topo*tcoef
    const float ah = arw[b * 256 + h];
    const float av = acl[b * 256 + w];
    const float at = ah * av;
    #pragma unroll
    for (int r = 0; r < 8; ++r) {
      const int m = m0 + r + half * 8;
      const size_t oi = (size_t)m * 65536 + (size_t)p;
      out[oi] = cox[m] * x[oi] + ah * accR[r] + av * accC[r] + at * tco[m * 256 + h];
    }
  }
}

// ---------------------------------------------------------------------------
// Launch wiring.  Workspace layout (floats):
//   Xn[65536] | attR[131072] | attC[131072] | tco[65536] | cox[256]
//   | arw[512] | acl[512] | Rbf(bf16,65536) | Cbf(bf16,65536)   (~1.84 MB)
// ---------------------------------------------------------------------------
extern "C" void kernel_launch(void* const* d_in, const int* in_sizes, int n_in,
                              void* d_out, int out_size, void* d_ws, size_t ws_size,
                              hipStream_t stream) {
  const float* x     = (const float*)d_in[0];
  const float* U     = (const float*)d_in[1];
  const float* S     = (const float*)d_in[2];
  const float* V     = (const float*)d_in[3];
  const float* G     = (const float*)d_in[4];
  const float* w_row = (const float*)d_in[5];
  const float* w_col = (const float*)d_in[6];
  const float* boast = (const float*)d_in[7];
  const float* res   = (const float*)d_in[8];
  const float* rdw   = (const float*)d_in[9];
  const float* gcw   = (const float*)d_in[10];
  const float* gcb   = (const float*)d_in[11];
  const float* exw   = (const float*)d_in[12];
  float* out = (float*)d_out;
  float* ws  = (float*)d_ws;

  float*  Xn   = ws;
  float*  attR = ws + 65536;
  float*  attC = ws + 196608;
  float*  tco  = ws + 327680;
  float*  cox  = ws + 393216;
  float*  arw  = ws + 393472;
  float*  acl  = ws + 393984;
  __bf16* Rbf  = (__bf16*)(ws + 394496);
  __bf16* Cbf  = (__bf16*)(ws + 394496 + 32768);

  k_rowmean<<<8192, 256, 0, stream>>>(x, Xn);
  k_small<<<2, 256, 0, stream>>>(U, S, V, w_row, w_col, boast, res, rdw, gcw,
                                 gcb, exw, Xn, attR, attC, tco, cox, arw, acl);
  k_R<<<8192, 256, 0, stream>>>(x, attR, Rbf);
  k_Cctx<<<256, 256, 0, stream>>>(x, attC, Cbf);
  k_gemm<<<dim3(4096, 4), 32, 0, stream>>>(G, x, Rbf, Cbf, cox, arw, acl, tco, out);
}
